// LinearCDE_17102559773094
// MI455X (gfx1250) — matrix operand order repaired and checked by reading, not compiled, not run on hardware
//
#include <hip/hip_runtime.h>
#include <hip/hip_bf16.h>

typedef __attribute__((ext_vector_type(16))) __bf16 v16bf;
typedef __attribute__((ext_vector_type(8)))  float  v8f;

#define SEQ    4096
#define DIN    64
#define H      128
#define NCH    64          // chunks
#define LCH    64          // steps per chunk (last chunk: 63 valid)
#define SB     8           // phase-1 sub-block length
#define TSCALE (1.0f/4095.0f)

static __device__ __forceinline__ v8f wmma_bf16(v16bf a, v16bf b, v8f c) {
  return __builtin_amdgcn_wmma_f32_16x16x32_bf16(false, a, false, b, (short)0, c, false, false);
}

// ---------- WMMA 16-bit operand swizzles (wave32, contiguous k = v + 16*half) ----------
// A tile (16 M x 32 K): lane = mi + 16*half; value v -> kk = v + 16*half
static __device__ __forceinline__ int a_kk(int v, int half) {
  return v + 16*half;
}
static __device__ __forceinline__ void a_inv(int mi, int kk, int& lane, int& v) {
  int half = kk >> 4;
  v = kk & 15;
  lane = mi + 16*half;
}
// B tile (32 K x 16 N): lane = n + 16*half; value v -> kk = v + 16*half
static __device__ __forceinline__ void b_inv(int kk, int n, int& lane, int& v) {
  lane = n + 16*(kk >> 4);
  v = kk & 15;
}

// =====================================================================
// Kernel 1: repack weights/activations into bf16 WMMA-native layouts.
//  Xb      [64][4096][64]         bf16 row-major
//  WxASw   [2][1024][32][16]      B-layout tiles of WxA^T (K=64, N=16384)
//  WflatSw [4][520][32][16]       B-layout tiles of W_flat (K=128, N=65*128)
//  WxBSw   [2][8][32][16]         B-layout tiles of WxB^T  (K=64, N=128)
//  Acbf    [128][128]             bf16 row-major (ts * W_A[:,0])
//  WoutSw  [4][4][32][16]         B-layout tiles of W_out^T (K=128, N=64)
// =====================================================================
__global__ void k_prep(const float* __restrict__ X, const float* __restrict__ W_A,
                       const float* __restrict__ W_B, const float* __restrict__ W_out,
                       __bf16* __restrict__ Xb, __bf16* __restrict__ WxASw,
                       __bf16* __restrict__ WflatSw, __bf16* __restrict__ WxBSw,
                       __bf16* __restrict__ Acbf, __bf16* __restrict__ WoutSw)
{
  size_t gid = (size_t)blockIdx.x * 256 + threadIdx.x;
  const size_t N_XB    = (size_t)64*4096*64;  // 16777216
  const size_t N_WXA   = (size_t)2*1024*512;  // 1048576
  const size_t N_WFLAT = (size_t)4*520*512;   // 1064960
  const size_t N_WXB   = (size_t)2*8*512;     // 8192
  const size_t N_AC    = 16384;
  const size_t N_WOUT  = (size_t)4*4*512;     // 8192

  if (gid < N_XB) { Xb[gid] = (__bf16)X[gid]; return; }
  gid -= N_XB;
  if (gid < N_WXA) {
    size_t idx = gid;
    int kc = (int)(idx >> 19);
    int rem = (int)(idx & ((1u<<19)-1));
    int nt = rem >> 9; int l = (rem >> 4) & 31; int v = rem & 15;
    int kk = v + 16*(l >> 4);
    int n  = nt*16 + (l & 15);
    int k  = kc*32 + kk;
    WxASw[gid] = (__bf16)W_A[(size_t)n*65 + k + 1];
    return;
  }
  gid -= N_WXA;
  if (gid < N_WFLAT) {
    size_t idx = gid;
    int jc = (int)(idx / (520*512));
    int rem = (int)(idx % (520*512));
    int nt = rem >> 9; int l = (rem >> 4) & 31; int v = rem & 15;
    int kk = v + 16*(l >> 4);
    int j  = jc*32 + kk;                       // K index (= y component)
    int n  = nt*16 + (l & 15);                 // N = k*128 + i
    int k = n >> 7, i = n & 127;
    float val = (k < 64) ? W_A[(size_t)(i*128 + j)*65 + k + 1]
                         : TSCALE * W_A[(size_t)(i*128 + j)*65];  // Ac slice
    WflatSw[gid] = (__bf16)val;
    return;
  }
  gid -= N_WFLAT;
  if (gid < N_WXB) {
    size_t idx = gid;
    int kc = (int)(idx / (8*512));
    int rem = (int)(idx % (8*512));
    int nt = rem >> 9; int l = (rem >> 4) & 31; int v = rem & 15;
    int kk = v + 16*(l >> 4);
    int h  = nt*16 + (l & 15);
    int k  = kc*32 + kk;
    WxBSw[gid] = (__bf16)W_B[(size_t)h*65 + k + 1];
    return;
  }
  gid -= N_WXB;
  if (gid < N_AC) { Acbf[gid] = (__bf16)(TSCALE * W_A[gid*65]); return; }
  gid -= N_AC;
  if (gid < N_WOUT) {
    size_t idx = gid;
    int jc = (int)(idx / (4*512));
    int rem = (int)(idx % (4*512));
    int nt = rem >> 9; int l = (rem >> 4) & 31; int v = rem & 15;
    int kk = v + 16*(l >> 4);
    int i  = jc*32 + kk;
    int o  = nt*16 + (l & 15);
    WoutSw[gid] = (__bf16)W_out[(size_t)o*128 + i];
  }
}

// =====================================================================
// Kernel 2: Bs[b][t][h] = bc + WxB @ x_t, all (b,t), via WMMA.
// Grid: 16384 M-tiles (rows m = b*4096 + t), 8 waves -> 8 N-tiles.
// =====================================================================
__global__ void k_bs(const __bf16* __restrict__ Xb, const __bf16* __restrict__ WxBSw,
                     const float* __restrict__ W_B, float* __restrict__ Bs)
{
  __shared__ __bf16 Ust[2*512];
  const int m0 = blockIdx.x * 16;
  const int tid = threadIdx.x, lane = tid & 31, w = tid >> 5;
  const int colin = lane & 15, rsel = (lane >> 4) * 8;

  for (int e = tid; e < 1024; e += 256) {
    int kc = e >> 9, l = (e >> 4) & 31, v = e & 15;
    int half = l >> 4, mi = l & 15;
    int k = kc*32 + a_kk(v, half);
    Ust[e] = Xb[(size_t)(m0 + mi)*64 + k];
  }
  __syncthreads();

  v16bf a0 = *(const v16bf*)(Ust + lane*16);
  v16bf a1 = *(const v16bf*)(Ust + 512 + lane*16);
  v16bf b0 = *(const v16bf*)(WxBSw + ((size_t)(0*8 + w))*512 + lane*16);
  v16bf b1 = *(const v16bf*)(WxBSw + ((size_t)(1*8 + w))*512 + lane*16);
  v8f z = {};
  z = wmma_bf16(a0, b0, z);
  z = wmma_bf16(a1, b1, z);

  int n = w*16 + colin;
  float bc = TSCALE * W_B[(size_t)n*65];
#pragma unroll
  for (int r = 0; r < 8; ++r) {
    int m = m0 + r + rsel;
    Bs[(size_t)m*128 + n] = z[r] + bc;
  }
}

// =====================================================================
// Kernel 3: phase 1 — per (chunk, batch) build the AUGMENTED transition
//   [P_c | q_c] as 9 column tiles (q is column tile 8, updated by the
//   same WMMA loop: q' = q + A_s q, then += B_t on colin==0 lanes).
// Grid 4096 blocks x 256 threads (8 waves). Dynamic LDS ~295 KB.
// =====================================================================
__global__ void k_phase1(const __bf16* __restrict__ Xb, const __bf16* __restrict__ WxASw,
                         const __bf16* __restrict__ Acbf, const float* __restrict__ Bs,
                         __bf16* __restrict__ Pc, float* __restrict__ qc)
{
  extern __shared__ char smem_raw[];
  __bf16* Abuf = (__bf16*)smem_raw;             // [SB][8 istrip][4 jc][512]
  __bf16* Pbuf = Abuf + (size_t)SB*8*4*512;     // [4 jc][9 nt][512]
  __bf16* Ust  = Pbuf + (size_t)4*9*512;        // [2 kc][512]
  __bf16* pad  = Ust + 2*512;                   // [256] scatter sink

  const int b = blockIdx.x & 63;
  const int c = blockIdx.x >> 6;
  const int tid = threadIdx.x, lane = tid & 31, w = tid >> 5;
  const int colin = lane & 15, rsel = (lane >> 4) * 8;
  const int rowbase = w * 16;
  const int t0c = 1 + c * LCH;

  v8f acc[9];                   // tiles 0..7: P row strip; tile 8: q column
#pragma unroll
  for (int ct = 0; ct < 8; ++ct) {
#pragma unroll
    for (int r = 0; r < 8; ++r) {
      int row = rowbase + r + rsel, col = ct*16 + colin;
      acc[ct][r] = (row == col) ? 1.0f : 0.0f;
    }
  }
#pragma unroll
  for (int r = 0; r < 8; ++r) acc[8][r] = 0.0f;

  for (int sb = 0; sb < LCH / SB; ++sb) {
    const int tbase = t0c + sb * SB;

    // stage U rows (A-layout, 16 rows: steps 0..7 valid, rest zero)
    for (int e = tid; e < 2*512; e += 256) {
      int kc = e >> 9, l = (e >> 4) & 31, v = e & 15;
      int half = l >> 4, mi = l & 15;
      int j = kc*32 + a_kk(v, half);
      int t = tbase + mi;
      __bf16 val = (__bf16)0.0f;
      if (mi < SB && t <= 4095) val = Xb[((size_t)b*SEQ + t)*64 + j];
      Ust[e] = val;
    }
    __syncthreads();

    // A-GEMM: [16x64] @ [64x16384]; wave w handles N-tiles w, w+8, ...
    v16bf ua0 = *(const v16bf*)(Ust + lane*16);
    v16bf ua1 = *(const v16bf*)(Ust + 512 + lane*16);
    for (int nt = w; nt < 1024; nt += 8) {
      if (nt + 8 < 1024) {
        __builtin_prefetch(WxASw + ((size_t)(nt + 8))*512 + lane*16, 0, 3);
        __builtin_prefetch(WxASw + ((size_t)(1024 + nt + 8))*512 + lane*16, 0, 3);
      }
      v16bf wb0 = *(const v16bf*)(WxASw + ((size_t)nt)*512 + lane*16);
      v16bf wb1 = *(const v16bf*)(WxASw + ((size_t)(1024 + nt))*512 + lane*16);
      v8f z = {};
      z = wmma_bf16(ua0, wb0, z);
      z = wmma_bf16(ua1, wb1, z);
      // scatter A_s rows (C-layout: VGPR r -> step r on lanes 0..15) + Ac
      int n0 = nt * 16;
      int i = n0 >> 7;
      int j = (n0 & 127) + colin;
      float ac = (float)Acbf[i*128 + j];
      int istrip = i >> 4, mi2 = i & 15;
      int jc = j >> 5, kk = j & 31;
      int dl, dv; a_inv(mi2, kk, dl, dv);
#pragma unroll
      for (int r = 0; r < 8; ++r) {
        __bf16* dst = (lane < 16)
            ? (Abuf + (((size_t)(r*8 + istrip)*4 + jc)*512 + dl*16 + dv))
            : (pad + (lane - 16)*8 + r);         // sink for padded rows
        *dst = (__bf16)(z[r] + ac);
      }
    }
    __syncthreads();

    // serial step updates within the sub-block
    for (int s = 0; s < SB; ++s) {
      int t = tbase + s;
      if (t > 4095) break;

      // stage current [P|q] into Pbuf (B-layout, bf16), 9 column tiles
#pragma unroll
      for (int ct = 0; ct < 9; ++ct) {
#pragma unroll
        for (int r = 0; r < 8; ++r) {
          int j = rowbase + r + rsel;
          int jc = j >> 5, kk = j & 31;
          int dl, dv; b_inv(kk, colin, dl, dv);
          Pbuf[((size_t)(jc*9 + ct))*512 + dl*16 + dv] = (__bf16)acc[ct][r];
        }
      }
      __syncthreads();

      // [P|q] += A_s @ [P|q] : 9 output tiles x 4 K-chunks of WMMA
      v16bf aop[4];
#pragma unroll
      for (int jt = 0; jt < 4; ++jt)
        aop[jt] = *(const v16bf*)(Abuf + (((size_t)(s*8 + w)*4 + jt)*512 + lane*16));
#pragma unroll
      for (int ct2 = 0; ct2 < 9; ++ct2) {
        v8f a2 = acc[ct2];
#pragma unroll
        for (int jt = 0; jt < 4; ++jt) {
          v16bf bop = *(const v16bf*)(Pbuf + ((size_t)(jt*9 + ct2))*512 + lane*16);
          a2 = wmma_bf16(aop[jt], bop, a2);
        }
        acc[ct2] = a2;
      }
      // q += B_t (column 0 of tile 8 lives on colin==0 lanes)
      if (colin == 0) {
        const float* bsrow = Bs + ((size_t)b*SEQ + t)*128;
#pragma unroll
        for (int r = 0; r < 8; ++r)
          acc[8][r] += bsrow[rowbase + r + rsel];
      }
      __syncthreads();
    }
  }

  // write back chunk transition
  size_t pbase = (size_t)(c*64 + b) * 16384;
#pragma unroll
  for (int ct = 0; ct < 8; ++ct) {
#pragma unroll
    for (int r = 0; r < 8; ++r) {
      int row = rowbase + r + rsel, col = ct*16 + colin;
      Pc[pbase + (size_t)row*128 + col] = (__bf16)acc[ct][r];
    }
  }
  if (colin == 0) {
#pragma unroll
    for (int r = 0; r < 8; ++r)
      qc[(size_t)(c*64 + b)*128 + rowbase + r + rsel] = acc[8][r];
  }
}

// =====================================================================
// Kernel 4: phase 2 — y0 and serial scan across 64 chunk boundaries.
// Grid: 64 blocks (one per batch), 128 threads.
// =====================================================================
__global__ void k_phase2(const float* __restrict__ X, const float* __restrict__ W_init,
                         const float* __restrict__ b_init, const __bf16* __restrict__ Pc,
                         const float* __restrict__ qc, float* __restrict__ ybound)
{
  __shared__ float y[128];
  __shared__ float yn[128];
  const int b = blockIdx.x, i = threadIdx.x;
  float s = b_init[i];
#pragma unroll 8
  for (int k = 0; k < 64; ++k) s += W_init[i*64 + k] * X[(size_t)b*SEQ*64 + k];
  y[i] = s;
  ybound[(size_t)b*128 + i] = s;                 // ybound[0][b]
  __syncthreads();
  for (int c = 0; c < NCH; ++c) {
    const __bf16* P = Pc + (size_t)(c*64 + b)*16384 + (size_t)i*128;
    float a = qc[(size_t)(c*64 + b)*128 + i];
#pragma unroll 4
    for (int j = 0; j < 128; ++j) a += (float)P[j] * y[j];
    yn[i] = a;
    __syncthreads();
    y[i] = yn[i];
    if (c < NCH - 1) ybound[((size_t)(c+1)*64 + b)*128 + i] = yn[i];
    __syncthreads();
  }
}

// =====================================================================
// Kernel 5: phase 3 — per (chunk, 16-batch group) reconstruct y_t and
// emit out = y @ Wout^T + bout fused. Bs blocks are double-buffered
// into LDS with global_load_async_to_lds_b128 (ASYNCcnt) so their
// latency hides behind the 65-channel WMMA loop.
// Grid 256 blocks x 256 threads.
// =====================================================================
#define P3_YM_OFF   0u
#define P3_YBF_OFF  8192u
#define P3_XST_OFF  12288u
#define P3_BS_OFF   16448u      // 2 x 8192 B double buffer
#define P3_LDS_SZ   (16448u + 16384u)

static __device__ __forceinline__ void issue_bs_async(const float* __restrict__ Bs,
    int bbase, int t, unsigned buf, int tid)
{
#pragma unroll
  for (int ss = 0; ss < 2; ++ss) {
    int seg = tid + ss*256;            // 512 x 16-byte segments
    int bb = seg >> 5, part = seg & 31;
    unsigned lds_off = P3_BS_OFF + buf*8192u + (unsigned)(bb*512 + part*16);
    unsigned long long g =
        (unsigned long long)(Bs + ((size_t)(bbase + bb)*SEQ + t)*128 + part*4);
    asm volatile("global_load_async_to_lds_b128 %0, %1, off"
                 :: "v"(lds_off), "v"(g) : "memory");
  }
}

static __device__ __forceinline__ void stage_y_x(const float* __restrict__ X,
    const float* __restrict__ Ym, __bf16* __restrict__ Ybf, float* __restrict__ xst,
    int bbase, int t, int tid)
{
  for (int e = tid; e < 2048 + 16*65; e += 256) {
    if (e < 2048) {
      int jc = e >> 9, l = (e >> 4) & 31, v = e & 15;
      int half = l >> 4, bb = l & 15;
      int j = jc*32 + a_kk(v, half);
      Ybf[e] = (__bf16)Ym[bb*128 + j];
    } else {
      int e2 = e - 2048;
      int bb = e2 / 65, k = e2 % 65;
      float val;
      if (k == 64) val = 1.0f;                  // folded Ac channel
      else val = (t <= 4095) ? X[(((size_t)(bbase + bb))*SEQ + t)*64 + k] : 0.0f;
      xst[bb*65 + k] = val;
    }
  }
}

static __device__ __forceinline__ void emit_out(const __bf16* __restrict__ Ybf,
    const __bf16* __restrict__ WoutSw, float* __restrict__ out,
    float boutv, int bbase, int s, int w, int lane)
{
  v16bf ya[4];
#pragma unroll
  for (int jc = 0; jc < 4; ++jc) ya[jc] = *(const v16bf*)(Ybf + jc*512 + lane*16);
  v8f oc = {};
#pragma unroll
  for (int jc = 0; jc < 4; ++jc) {
    v16bf bop = *(const v16bf*)(WoutSw + ((size_t)(jc*4 + w))*512 + lane*16);
    oc = wmma_bf16(ya[jc], bop, oc);
  }
  int colin = lane & 15, rsel = (lane >> 4) * 8;
#pragma unroll
  for (int r = 0; r < 8; ++r) {
    int bb = bbase + r + rsel;
    out[((size_t)bb*SEQ + s)*64 + w*16 + colin] = oc[r] + boutv;
  }
}

__global__ void k_phase3(const float* __restrict__ X, const __bf16* __restrict__ WflatSw,
                         const __bf16* __restrict__ WoutSw, const float* __restrict__ Bs,
                         const float* __restrict__ ybound, const float* __restrict__ b_out,
                         float* __restrict__ out)
{
  extern __shared__ char smem_raw[];
  float*  Ym    = (float*)(smem_raw + P3_YM_OFF);   // [16][128]
  __bf16* Ybf   = (__bf16*)(smem_raw + P3_YBF_OFF); // [4 jc][512] A-layout
  float*  xst   = (float*)(smem_raw + P3_XST_OFF);  // [16][65]
  float*  BsBuf = (float*)(smem_raw + P3_BS_OFF);   // [2][16][128]

  const int c = blockIdx.x >> 2;
  const int g = blockIdx.x & 3;
  const int bbase = g * 16;
  const int tid = threadIdx.x, lane = tid & 31, w = tid >> 5;
  const int colin = lane & 15, rsel = (lane >> 4) * 8;
  const int t0 = c * LCH + 1;

  for (int e = tid; e < 2048; e += 256) {
    int bb = e >> 7, i = e & 127;
    Ym[e] = ybound[((size_t)c*64 + bbase + bb)*128 + i];
  }
  issue_bs_async(Bs, bbase, t0, 0, tid);          // prefetch B_{t0} -> buf 0
  __syncthreads();
  stage_y_x(X, Ym, Ybf, xst, bbase, t0, tid);
  __syncthreads();

  float boutv = (w < 4) ? b_out[w*16 + colin] : 0.0f;
  if (c == 0 && w < 4) emit_out(Ybf, WoutSw, out, boutv, bbase, 0, w, lane);

  for (int s = 0; s < LCH; ++s) {
    const int t = t0 + s;
    if (t > 4095) break;
    const unsigned cur = (unsigned)(s & 1);

    // async copy of next step's Bs block overlaps this step's WMMAs
    if (t + 1 <= 4095) issue_bs_async(Bs, bbase, t + 1, cur ^ 1u, tid);

    // A-operand: Y (y_{t-1}) for all 4 K-chunks
    v16bf ya[4];
#pragma unroll
    for (int jc = 0; jc < 4; ++jc) ya[jc] = *(const v16bf*)(Ybf + jc*512 + lane*16);

    // AY strip for columns [16w,16w+16): sum over 65 channels of x_k * T_k
    v8f ay = {};
    for (int k = 0; k < 65; ++k) {
      int nt = k*8 + w;
      if (k < 64)
        __builtin_prefetch(WflatSw + ((size_t)(k*8 + 8 + w))*512 + lane*16, 0, 3);
      v8f tk = {};
#pragma unroll
      for (int jc = 0; jc < 4; ++jc) {
        v16bf bop = *(const v16bf*)(WflatSw + ((size_t)(jc*520 + nt))*512 + lane*16);
        tk = wmma_bf16(ya[jc], bop, tk);
      }
#pragma unroll
      for (int r = 0; r < 8; ++r)
        ay[r] += tk[r] * xst[(r + rsel)*65 + k];
    }

    // drain older async ops: keep only the 2 just-issued outstanding
    if (t + 1 <= 4095)
      asm volatile("s_wait_asynccnt 0x2" ::: "memory");
    else
      asm volatile("s_wait_asynccnt 0x0" ::: "memory");
    __syncthreads();   // make every wave's async LDS writes visible

    // y_t = y_{t-1} + AY + B_t  (wave-private column strip, no conflicts)
#pragma unroll
    for (int r = 0; r < 8; ++r) {
      int bb = r + rsel;
      int i = w*16 + colin;
      float yv = Ym[bb*128 + i] + ay[r] + BsBuf[cur*2048 + bb*128 + i];
      Ym[bb*128 + i] = yv;
    }
    __syncthreads();
    stage_y_x(X, Ym, Ybf, xst, bbase, t + 1, tid);
    __syncthreads();

    // fused output projection for row s = t (waves 0..3, wave-uniform)
    if (w < 4) emit_out(Ybf, WoutSw, out, boutv, bbase, t, w, lane);
  }
}

// =====================================================================
extern "C" void kernel_launch(void* const* d_in, const int* in_sizes, int n_in,
                              void* d_out, int out_size, void* d_ws, size_t ws_size,
                              hipStream_t stream) {
  const float* X      = (const float*)d_in[0];
  const float* W_init = (const float*)d_in[1];
  const float* b_init = (const float*)d_in[2];
  const float* W_A    = (const float*)d_in[3];
  const float* W_B    = (const float*)d_in[4];
  const float* W_out  = (const float*)d_in[5];
  const float* b_out  = (const float*)d_in[6];
  float* out = (float*)d_out;

  // workspace partition (~311 MB)
  char* ws = (char*)d_ws;
  size_t o = 0;
  auto alloc = [&](size_t bytes) -> void* {
    o = (o + 255) & ~(size_t)255;
    void* p = ws + o;
    o += bytes;
    return p;
  };
  __bf16* Xb      = (__bf16*)alloc((size_t)64*4096*64 * 2);
  __bf16* WxASw   = (__bf16*)alloc((size_t)2*1024*512 * 2);
  __bf16* WflatSw = (__bf16*)alloc((size_t)4*520*512 * 2);
  __bf16* WxBSw   = (__bf16*)alloc((size_t)2*8*512 * 2);
  __bf16* Acbf    = (__bf16*)alloc((size_t)16384 * 2);
  __bf16* WoutSw  = (__bf16*)alloc((size_t)4*4*512 * 2);
  float*  Bs      = (float*)alloc((size_t)64*4096*128 * 4);
  __bf16* Pc      = (__bf16*)alloc((size_t)64*64*16384 * 2);
  float*  qc      = (float*)alloc((size_t)64*64*128 * 4);
  float*  ybound  = (float*)alloc((size_t)64*64*128 * 4);

  const size_t PREP_TOTAL = 16777216ULL + 1048576 + 1064960 + 8192 + 16384 + 8192;
  const int prep_blocks = (int)((PREP_TOTAL + 255) / 256);

  k_prep<<<prep_blocks, 256, 0, stream>>>(X, W_A, W_B, W_out,
                                          Xb, WxASw, WflatSw, WxBSw, Acbf, WoutSw);
  k_bs<<<16384, 256, 0, stream>>>(Xb, WxBSw, W_B, Bs);

  const size_t P1_LDS = (size_t)SB*8*4*512*2   // Abuf   262144
                      + (size_t)4*9*512*2      // Pbuf    36864
                      + (size_t)2*512*2        // Ust      2048
                      + 256*2;                 // pad       512
  k_phase1<<<NCH*64, 256, P1_LDS, stream>>>(Xb, WxASw, Acbf, Bs, Pc, qc);

  k_phase2<<<64, 128, 0, stream>>>(X, W_init, b_init, Pc, qc, ybound);

  k_phase3<<<NCH*4, 256, P3_LDS_SZ, stream>>>(X, WflatSw, WoutSw, Bs, ybound, b_out, out);
}
